// MHAttention_83794811945503
// MI455X (gfx1250) — matrix operand order repaired and checked by reading, not compiled, not run on hardware
//
#include <hip/hip_runtime.h>
#include <hip/hip_bf16.h>

#define B_  2
#define T_  2048
#define C_  1024
#define H_  16
#define CA_ 64
#define BT_ (B_ * T_)

typedef __bf16 bf16;
typedef __attribute__((ext_vector_type(16))) __bf16 v16bf;
typedef __attribute__((ext_vector_type(8)))  __bf16 v8bf;
typedef __attribute__((ext_vector_type(8)))  float  v8f;
typedef __attribute__((ext_vector_type(4)))  int    v4i;

#if __has_builtin(__builtin_amdgcn_global_load_async_to_lds_b128)
#define HAVE_ASYNC_LDS 1
#else
#define HAVE_ASYNC_LDS 0
#endif

// builtin wants v4i pointers in explicit address spaces:
//   arg0: global (addrspace(1)), arg1: LDS (addrspace(3))
#define GAS128(p) ((__attribute__((address_space(1))) v4i*)(p))
#define LAS128(p) ((__attribute__((address_space(3))) v4i*)(p))

// ---- fragment helpers ------------------------------------------------------

static __device__ __forceinline__ v16bf load16(const bf16* p) {
  union { v16bf v; v8bf h[2]; } u;
  u.h[0] = *(const v8bf*)(p);
  u.h[1] = *(const v8bf*)(p + 8);
  return u.v;
}

// A operand (16x32 bf16, M x K), row-major source with leading dim ld.
// ISA layout: lane<16 holds row M=lane, K=0..7 (v0-3) and K=16..23 (v4-7);
// lane>=16 holds same row, K=8..15 and K=24..31.
static __device__ __forceinline__ v16bf loadA(const bf16* base, int ld, int lane) {
  const int row = lane & 15;
  const int kh  = (lane >> 4) * 8;
  const bf16* p = base + row * ld + kh;
  union { v16bf v; v8bf h[2]; } u;
  u.h[0] = *(const v8bf*)(p);        // K = kh .. kh+7
  u.h[1] = *(const v8bf*)(p + 16);   // K = 16+kh .. 16+kh+7
  return u.v;
}

// B operand (32x16 bf16, K x N), row-major source: lane holds column N=lane%16,
// K in the same order as loadA (kh..kh+7, 16+kh..16+kh+7).
static __device__ __forceinline__ v16bf loadB(const bf16* base, int ld, int lane) {
  const int n  = lane & 15;
  const int kh = (lane >> 4) * 8;
  v16bf v;
  #pragma unroll
  for (int s = 0; s < 8; ++s) {
    v[s]     = base[(kh + s) * ld + n];
    v[8 + s] = base[(16 + kh + s) * ld + n];
  }
  return v;
}

// reductions across the 16 lanes holding one C/D row (masks < 16 stay in half)
static __device__ __forceinline__ float hmax16(float v) {
  #pragma unroll
  for (int m = 1; m < 16; m <<= 1) v = fmaxf(v, __shfl_xor(v, m, 32));
  return v;
}
static __device__ __forceinline__ float hsum16(float v) {
  #pragma unroll
  for (int m = 1; m < 16; m <<= 1) v += __shfl_xor(v, m, 32);
  return v;
}

#define WMMA_BF16(a, b, c) \
  __builtin_amdgcn_wmma_f32_16x16x32_bf16(false, (a), false, (b), (short)0, (c), false, false)

// ---- kernel 0: f32 -> bf16 -------------------------------------------------

__global__ void cvt_f32_to_bf16(const float* __restrict__ in,
                                bf16* __restrict__ out, int n4) {
  int i = blockIdx.x * blockDim.x + threadIdx.x;
  if (i < n4) {
    float4 f = ((const float4*)in)[i];
    union { bf16 b[4]; unsigned long long u; } o;
    o.b[0] = (bf16)f.x; o.b[1] = (bf16)f.y;
    o.b[2] = (bf16)f.z; o.b[3] = (bf16)f.w;
    ((unsigned long long*)out)[i] = o.u;
  }
}

// ---- kernel 1: QKV projection (K stored transposed) ------------------------
// grid: (BT/128, H, 3), block 256 (8 waves). Wave computes a 16x64 tile.

__global__ __launch_bounds__(256) void qkv_proj(
    const bf16* __restrict__ Xb,                       // [BT, C]
    const bf16* __restrict__ Wq, const bf16* __restrict__ Wk,
    const bf16* __restrict__ Wv,                       // [H, C, Ca]
    bf16* __restrict__ Qb,                             // [B,H,T,Ca]
    bf16* __restrict__ Ktb,                            // [B,H,Ca,T]
    bf16* __restrict__ Vb)                             // [B,H,T,Ca]
{
  const int wave = threadIdx.x >> 5;
  const int lane = threadIdx.x & 31;
  const int h = blockIdx.y;
  const int z = blockIdx.z;
  const int r0 = blockIdx.x * 128 + wave * 16;         // row in [0, BT)

  const bf16* W = (z == 0) ? Wq : (z == 1) ? Wk : Wv;
  W += (size_t)h * C_ * CA_;

  v8f acc[4];
  #pragma unroll
  for (int t = 0; t < 4; ++t)
    #pragma unroll
    for (int j = 0; j < 8; ++j) acc[t][j] = 0.0f;

  for (int kk = 0; kk < C_; kk += 32) {
    if (kk + 64 < C_) {  // prefetch next-next K-slab, WGP scope (near caches)
      __builtin_prefetch(Xb + (size_t)r0 * C_ + kk + 64, 0, 3);
      __builtin_prefetch(W + (size_t)(kk + 64) * CA_, 0, 3);
    }
    v16bf a = loadA(Xb + (size_t)r0 * C_ + kk, C_, lane);
    #pragma unroll
    for (int t = 0; t < 4; ++t) {
      v16bf b = loadB(W + (size_t)kk * CA_ + 16 * t, CA_, lane);
      acc[t] = WMMA_BF16(a, b, acc[t]);
    }
  }

  const int lh = lane >> 4, col = lane & 15;
  const int bidx = r0 / T_, t0 = r0 % T_;
  #pragma unroll
  for (int t = 0; t < 4; ++t) {
    #pragma unroll
    for (int j = 0; j < 8; ++j) {
      const int row = t0 + j + 8 * lh;
      const int a   = 16 * t + col;
      const bf16 v = (bf16)acc[t][j];
      if (z == 1)      Ktb[(((size_t)bidx * H_ + h) * CA_ + a) * T_ + row] = v;
      else if (z == 0) Qb [(((size_t)bidx * H_ + h) * T_ + row) * CA_ + a] = v;
      else             Vb [(((size_t)bidx * H_ + h) * T_ + row) * CA_ + a] = v;
    }
  }
}

// ---- kernel 2: causal flash attention --------------------------------------
// grid: (T/128, H, B), block 256. Wave w owns query rows qBlock + 16w .. +15.
// K/V tiles double-buffered in LDS, staged with async global->LDS b128 copies
// tracked by ASYNCcnt so tile i+1 streams in while tile i feeds the WMMAs.

__global__ __launch_bounds__(256) void flash_attn(
    const bf16* __restrict__ Qb,   // [B,H,T,Ca]
    const bf16* __restrict__ Ktb,  // [B,H,Ca,T]
    const bf16* __restrict__ Vb,   // [B,H,T,Ca]
    bf16* __restrict__ Yb)         // [B,T,C]  (heads concatenated)
{
  __shared__ bf16 ktile[2][64 * 64];     // K^T tile: [Ca=64][64 keys]
  __shared__ bf16 vtile[2][64 * 64];     // V tile:   [64 keys][Ca=64]
  __shared__ bf16 pbuf[8][16 * 32];      // per-wave P staging (C->A relayout)

  const int tid  = threadIdx.x;
  const int wave = tid >> 5, lane = tid & 31;
  const int lh   = lane >> 4, col = lane & 15;
  const int h = blockIdx.y, b = blockIdx.z;
  const int qBlock = blockIdx.x * 128;
  const int qRow0  = qBlock + wave * 16;
  const float scale = 0.125f;            // 1/sqrt(64)

  const bf16* Qh  = Qb  + ((size_t)(b * H_ + h) * T_) * CA_;
  const bf16* Kth = Ktb + ((size_t)(b * H_ + h) * CA_) * T_;
  const bf16* Vh  = Vb  + ((size_t)(b * H_ + h) * T_) * CA_;

  const v16bf qa0 = loadA(Qh + (size_t)qRow0 * CA_ +  0, CA_, lane);
  const v16bf qa1 = loadA(Qh + (size_t)qRow0 * CA_ + 32, CA_, lane);

  v8f o[4];
  float m[8], l[8];
  #pragma unroll
  for (int t = 0; t < 4; ++t)
    #pragma unroll
    for (int j = 0; j < 8; ++j) o[t][j] = 0.0f;
  #pragma unroll
  for (int j = 0; j < 8; ++j) { m[j] = -1e30f; l[j] = 0.0f; }

  // staging geometry: 256 threads x 32B cover each 8KB tile
  const int sr  = tid >> 2;              // 0..63
  const int sc0 = (tid & 3) * 16;        // 0,16,32,48
  const bf16* kSrc = Kth + (size_t)sr * T_ + sc0;   // + it*64
  const bf16* vSrc = Vh + (size_t)sr * CA_ + sc0;   // + it*64*CA_

  auto stage = [&](int it) {
    bf16* kd = &ktile[it & 1][sr * 64 + sc0];
    bf16* vd = &vtile[it & 1][sr * 64 + sc0];
    const bf16* ks = kSrc + it * 64;
    const bf16* vs = vSrc + (size_t)(it * 64) * CA_;
#if HAVE_ASYNC_LDS
    __builtin_amdgcn_global_load_async_to_lds_b128(GAS128(ks),     LAS128(kd),     0, 0);
    __builtin_amdgcn_global_load_async_to_lds_b128(GAS128(ks + 8), LAS128(kd + 8), 0, 0);
    __builtin_amdgcn_global_load_async_to_lds_b128(GAS128(vs),     LAS128(vd),     0, 0);
    __builtin_amdgcn_global_load_async_to_lds_b128(GAS128(vs + 8), LAS128(vd + 8), 0, 0);
#else
    *(v8bf*)(kd)     = *(const v8bf*)(ks);
    *(v8bf*)(kd + 8) = *(const v8bf*)(ks + 8);
    *(v8bf*)(vd)     = *(const v8bf*)(vs);
    *(v8bf*)(vd + 8) = *(const v8bf*)(vs + 8);
#endif
  };

  const int nT = (qBlock + 128) >> 6;    // causal: keys needed < qBlock+128
  stage(0);

  for (int it = 0; it < nT; ++it) {
    if (it + 1 < nT) {
      stage(it + 1);                                   // 4 async ops in flight
      asm volatile("s_wait_asynccnt 4" ::: "memory");  // tile `it` landed
    } else {
      asm volatile("s_wait_asynccnt 0" ::: "memory");
    }
    __syncthreads();

    const bf16* kt = &ktile[it & 1][0];
    const bf16* vt = &vtile[it & 1][0];
    const int sBase = it * 64;

    #pragma unroll
    for (int sb = 0; sb < 64; sb += 32) {
      const int s0 = sBase + sb;
      if (s0 <= qRow0 + 15) {            // wave-uniform: EXEC stays full
        // ---- S = Q * K^T  (16 rows x 32 keys) ----
        v8f sc0v, sc1v;
        #pragma unroll
        for (int j = 0; j < 8; ++j) { sc0v[j] = 0.0f; sc1v[j] = 0.0f; }
        #pragma unroll
        for (int ka = 0; ka < 64; ka += 32) {
          const v16bf a  = (ka == 0) ? qa0 : qa1;
          const v16bf b0 = loadB(&kt[ka * 64 + sb], 64, lane);
          const v16bf b1 = loadB(&kt[ka * 64 + sb + 16], 64, lane);
          sc0v = WMMA_BF16(a, b0, sc0v);
          sc1v = WMMA_BF16(a, b1, sc1v);
        }
        // ---- causal mask, scale, online softmax ----
        float p0[8], p1[8], corr[8];
        #pragma unroll
        for (int j = 0; j < 8; ++j) {
          const int qr = qRow0 + j + 8 * lh;
          const float v0 = (s0 + col      <= qr) ? sc0v[j] * scale : -1e30f;
          const float v1 = (s0 + 16 + col <= qr) ? sc1v[j] * scale : -1e30f;
          const float rm = hmax16(fmaxf(v0, v1));
          const float mn = fmaxf(m[j], rm);
          corr[j] = __expf(m[j] - mn);
          p0[j]   = __expf(v0 - mn);
          p1[j]   = __expf(v1 - mn);
          const float rs = hsum16(p0[j] + p1[j]);
          l[j] = l[j] * corr[j] + rs;
          m[j] = mn;
        }
        #pragma unroll
        for (int t = 0; t < 4; ++t)
          #pragma unroll
          for (int j = 0; j < 8; ++j) o[t][j] *= corr[j];

        // ---- relayout P (C-layout) -> A-layout via per-wave LDS ----
        bf16* pw = &pbuf[wave][0];
        #pragma unroll
        for (int j = 0; j < 8; ++j) {
          const int row = j + 8 * lh;
          pw[row * 32 + col]      = (bf16)p0[j];
          pw[row * 32 + 16 + col] = (bf16)p1[j];
        }
        asm volatile("s_wait_dscnt 0" ::: "memory");
        const v16bf pa = loadA(pw, 32, lane);

        // ---- O += P * V ----
        #pragma unroll
        for (int t = 0; t < 4; ++t) {
          const v16bf bv = loadB(&vt[sb * 64 + 16 * t], 64, lane);
          o[t] = WMMA_BF16(pa, bv, o[t]);
        }
      }
    }
    __syncthreads();
  }

  // epilogue: normalize and write head slice of Y (bf16)
  #pragma unroll
  for (int j = 0; j < 8; ++j) {
    const int qr = qRow0 + j + 8 * lh;
    const float inv = 1.0f / l[j];
    #pragma unroll
    for (int t = 0; t < 4; ++t)
      Yb[((size_t)(b * T_) + qr) * C_ + h * CA_ + 16 * t + col] = (bf16)(o[t][j] * inv);
  }
}

// ---- kernel 3: output projection (f32 out) ---------------------------------
// grid: (BT/128, C/64), block 256.

__global__ __launch_bounds__(256) void out_proj(
    const bf16* __restrict__ Yb,   // [BT, C]
    const bf16* __restrict__ Wob,  // [C, C]
    float* __restrict__ Out)       // [BT, C]
{
  const int wave = threadIdx.x >> 5, lane = threadIdx.x & 31;
  const int r0 = blockIdx.x * 128 + wave * 16;
  const int n0 = blockIdx.y * 64;

  v8f acc[4];
  #pragma unroll
  for (int t = 0; t < 4; ++t)
    #pragma unroll
    for (int j = 0; j < 8; ++j) acc[t][j] = 0.0f;

  for (int kk = 0; kk < C_; kk += 32) {
    if (kk + 64 < C_) {
      __builtin_prefetch(Yb + (size_t)r0 * C_ + kk + 64, 0, 3);
      __builtin_prefetch(Wob + (size_t)(kk + 64) * C_ + n0, 0, 3);
    }
    v16bf a = loadA(Yb + (size_t)r0 * C_ + kk, C_, lane);
    #pragma unroll
    for (int t = 0; t < 4; ++t) {
      v16bf b = loadB(Wob + (size_t)kk * C_ + n0 + 16 * t, C_, lane);
      acc[t] = WMMA_BF16(a, b, acc[t]);
    }
  }

  const int lh = lane >> 4, col = lane & 15;
  #pragma unroll
  for (int t = 0; t < 4; ++t)
    #pragma unroll
    for (int j = 0; j < 8; ++j)
      Out[(size_t)(r0 + j + 8 * lh) * C_ + n0 + 16 * t + col] = acc[t][j];
}

// ---- launcher --------------------------------------------------------------

extern "C" void kernel_launch(void* const* d_in, const int* in_sizes, int n_in,
                              void* d_out, int out_size, void* d_ws, size_t ws_size,
                              hipStream_t stream) {
  (void)in_sizes; (void)n_in; (void)out_size; (void)ws_size;
  const float* x  = (const float*)d_in[0];
  const float* wq = (const float*)d_in[1];
  const float* wk = (const float*)d_in[2];
  const float* wv = (const float*)d_in[3];
  const float* wo = (const float*)d_in[4];
  float* out = (float*)d_out;

  size_t off = 0;
  auto carve = [&](size_t elems) -> bf16* {
    bf16* p = (bf16*)((char*)d_ws + off);
    off += ((elems * sizeof(bf16)) + 255) & ~(size_t)255;
    return p;
  };
  bf16* Xb  = carve((size_t)BT_ * C_);
  bf16* Wqb = carve((size_t)H_ * C_ * CA_);
  bf16* Wkb = carve((size_t)H_ * C_ * CA_);
  bf16* Wvb = carve((size_t)H_ * C_ * CA_);
  bf16* Wob = carve((size_t)C_ * C_);
  bf16* Qb  = carve((size_t)B_ * H_ * T_ * CA_);
  bf16* Ktb = carve((size_t)B_ * H_ * T_ * CA_);
  bf16* Vb  = carve((size_t)B_ * H_ * T_ * CA_);
  bf16* Yb  = carve((size_t)BT_ * C_);

  auto cvt = [&](const float* src, bf16* dst, size_t n) {
    int n4 = (int)(n / 4);
    cvt_f32_to_bf16<<<(n4 + 255) / 256, 256, 0, stream>>>(src, dst, n4);
  };
  cvt(x,  Xb,  (size_t)BT_ * C_);
  cvt(wq, Wqb, (size_t)H_ * C_ * CA_);
  cvt(wk, Wkb, (size_t)H_ * C_ * CA_);
  cvt(wv, Wvb, (size_t)H_ * C_ * CA_);
  cvt(wo, Wob, (size_t)C_ * C_);

  qkv_proj<<<dim3(BT_ / 128, H_, 3), 256, 0, stream>>>(Xb, Wqb, Wkb, Wvb, Qb, Ktb, Vb);
  flash_attn<<<dim3(T_ / 128, H_, B_), 256, 0, stream>>>(Qb, Ktb, Vb, Yb);
  out_proj<<<dim3(BT_ / 128, C_ / 64), 256, 0, stream>>>(Yb, Wob, out);
}
